// AttentionHead_18262200943329
// MI455X (gfx1250) — matrix operand order repaired and checked by reading, not compiled, not run on hardware
//
#include <hip/hip_runtime.h>

typedef _Float16 v16h __attribute__((ext_vector_type(16)));
typedef _Float16 v8h  __attribute__((ext_vector_type(8)));
typedef _Float16 v4h  __attribute__((ext_vector_type(4)));
typedef float    v8f  __attribute__((ext_vector_type(8)));
typedef float    v4f  __attribute__((ext_vector_type(4)));

constexpr int kB    = 4;
constexpr int kS    = 2048;
constexpr int kDIN  = 1024;
constexpr int kDOUT = 128;

// ---------------- fragment load helper (two 16B segments -> v16h) ----------
static __device__ __forceinline__ v16h ldfrag(const _Float16* p0, const _Float16* p1) {
    v8h a = *(const v8h*)p0;
    v8h b = *(const v8h*)p1;
    v16h r;
#pragma unroll
    for (int i = 0; i < 8; ++i) { r[i] = a[i]; r[i + 8] = b[i]; }
    return r;
}

static __device__ __forceinline__ v8f wmma16(v16h a, v16h b, v8f c) {
    return __builtin_amdgcn_wmma_f32_16x16x32_f16(false, a, false, b, (short)0, c,
                                                  false, false);
}

// ===========================================================================
// Kernel 1: projection GEMM  X[8192,1024] @ W[1024,128] -> f16
//   blockIdx.y: 0 = K (row major), 1 = V (stored transposed [e][s]), 2 = Q
//   block: 128 threads (4 waves), each wave owns 16 rows, BM=64 rows/block
// ===========================================================================
constexpr int BM   = 64;
constexpr int KC   = 128;          // K-chunk staged per iteration
constexpr int AST  = KC + 8;       // LDS row strides (pad 16B -> conflict-free)
constexpr int BST  = KC + 8;

__global__ __launch_bounds__(128) void proj_kernel(
    const float* __restrict__ Xk, const float* __restrict__ Xv, const float* __restrict__ Xq,
    const float* __restrict__ Wk, const float* __restrict__ Wv, const float* __restrict__ Wq,
    _Float16* __restrict__ Kh, _Float16* __restrict__ Vt, _Float16* __restrict__ Qh)
{
    __shared__ _Float16 As[BM * AST];     // [64][136] f16 (row-major, m x k)
    __shared__ _Float16 Bs[kDOUT * BST];  // [128][136] f16 (W transposed: n x k)

    const int p = blockIdx.y;
    const float* X = (p == 0) ? Xk : (p == 1) ? Xv : Xq;
    const float* W = (p == 0) ? Wk : (p == 1) ? Wv : Wq;

    const int tid  = threadIdx.x;
    const int wave = tid >> 5;
    const int lane = tid & 31;
    const int lrow = lane & 15;
    const int ahi  = (lane & 16) ? 16 : 0;  // A-frag K segment select
    const int bhi  = (lane & 16) ? 16 : 0;  // B-frag K segment select
    const int rhi  = (lane & 16) ? 8 : 0;   // C/D row offset for upper lanes
    const int m0   = blockIdx.x * BM;

    v8f acc[8];
#pragma unroll
    for (int i = 0; i < 8; ++i) acc[i] = (v8f){};

    for (int kc = 0; kc < kDIN / KC; ++kc) {
        // ---- prefetch next X chunk (HBM streaming traffic) ----
        if (kc + 1 < kDIN / KC) {
            // 64 rows x 512B; 128 threads x 2 prefetches cover it
            int prow = tid >> 1;
            int pcol = (tid & 1) * 64;
            __builtin_prefetch(X + (size_t)(m0 + prow) * kDIN + (kc + 1) * KC + pcol, 0, 0);
            __builtin_prefetch(X + (size_t)(m0 + prow) * kDIN + (kc + 1) * KC + pcol + 32, 0, 0);
        }
        // ---- stage A: 64x128 f32 -> f16, row major ----
        for (int i = tid; i < BM * (KC / 4); i += 128) {
            int row = i >> 5;
            int c4  = (i & 31) * 4;
            v4f x = *(const v4f*)(X + (size_t)(m0 + row) * kDIN + kc * KC + c4);
            v4h h;
            h[0] = (_Float16)x[0]; h[1] = (_Float16)x[1];
            h[2] = (_Float16)x[2]; h[3] = (_Float16)x[3];
            *(v4h*)&As[row * AST + c4] = h;
        }
        // ---- stage B: 128(k)x128(n) f32, transpose -> Bs[n][k] f16 ----
        for (int i = tid; i < KC * (kDOUT / 4); i += 128) {
            int k  = i >> 5;
            int n4 = (i & 31) * 4;
            v4f w = *(const v4f*)(W + (size_t)(kc * KC + k) * kDOUT + n4);
#pragma unroll
            for (int j = 0; j < 4; ++j)
                Bs[(n4 + j) * BST + k] = (_Float16)w[j];
        }
        __syncthreads();

        const _Float16* Abase = &As[(wave * 16 + lrow) * AST];
#pragma unroll
        for (int ks = 0; ks < 4; ++ks) {
            const _Float16* ap = Abase + ks * 32 + ahi;
            v16h af = ldfrag(ap, ap + 8);
            // preload ALL B fragments first (distinct regs -> pipelined ds loads)
            v16h bf[8];
#pragma unroll
            for (int nt = 0; nt < 8; ++nt) {
                const _Float16* bp = &Bs[(nt * 16 + lrow) * BST + ks * 32 + bhi];
                bf[nt] = ldfrag(bp, bp + 8);
            }
#pragma unroll
            for (int nt = 0; nt < 8; ++nt)
                acc[nt] = wmma16(af, bf[nt], acc[nt]);
        }
        __syncthreads();
    }

    // ---- write back (C/D layout: VGPR r -> row r / r+8, col = lane&15) ----
    const int mrow = m0 + wave * 16;
    if (p == 1) {
        // V transposed: Vt[b][e][s]
#pragma unroll
        for (int nt = 0; nt < 8; ++nt) {
            int e = nt * 16 + lrow;
#pragma unroll
            for (int r = 0; r < 8; ++r) {
                int m  = mrow + r + rhi;
                int bi = m >> 11, s = m & (kS - 1);
                Vt[((size_t)bi * kDOUT + e) * kS + s] = (_Float16)acc[nt][r];
            }
        }
    } else {
        _Float16* O = (p == 0) ? Kh : Qh;
#pragma unroll
        for (int nt = 0; nt < 8; ++nt)
#pragma unroll
            for (int r = 0; r < 8; ++r) {
                int m = mrow + r + rhi;
                O[(size_t)m * kDOUT + nt * 16 + lrow] = (_Float16)acc[nt][r];
            }
    }
}

// ===========================================================================
// Kernel 2: causal flash attention. 1 wave = 16 query rows, full d=128.
//   scores = (Q Kt) * 1/sqrt(S)  (log2e folded in, softmax in base-2)
// ===========================================================================
__global__ __launch_bounds__(128) void attn_kernel(
    const _Float16* __restrict__ Qh, const _Float16* __restrict__ Kh,
    const _Float16* __restrict__ Vt, float* __restrict__ Out)
{
    __shared__ _Float16 Pl[4][16][40];   // per-wave P bounce (16 q x 32 k, pad 8)

    const int tid   = threadIdx.x;
    const int wave  = tid >> 5;
    const int lane  = tid & 31;
    const int lrow  = lane & 15;
    const int seg8  = (lane & 16) ? 16 : 0;
    const int seg16 = (lane & 16) ? 16 : 0;
    const int rhi   = (lane & 16) ? 8 : 0;

    const int qt = blockIdx.x * 4 + wave;    // global 16-row query tile id
    const int b  = qt >> 7;                  // kS/16 = 128 tiles per batch
    const int q0 = (qt & 127) << 4;

    const _Float16* Qb = Qh + ((size_t)b * kS + q0) * kDOUT;
    const _Float16* Kb = Kh + (size_t)b * kS * kDOUT;
    const _Float16* Vb = Vt + (size_t)b * kDOUT * kS;

    // preload Q A-fragments for all 4 e-steps (row = lane&15)
    v16h qf[4];
    {
        const _Float16* qr = Qb + (size_t)lrow * kDOUT;
#pragma unroll
        for (int es = 0; es < 4; ++es) {
            const _Float16* p = qr + es * 32 + seg8;
            qf[es] = ldfrag(p, p + 8);
        }
    }

    v8f o[8];
    float m_run[8], l_run[8];
#pragma unroll
    for (int i = 0; i < 8; ++i) { o[i] = (v8f){}; m_run[i] = -1e30f; l_run[i] = 0.f; }

    // 1/sqrt(2048) * log2(e): base-2 softmax == base-e softmax of scaled scores
    const float SCL = 0.022097086912079608f * 1.4426950408889634f;

    const int nkt = (q0 >> 5) + 1;           // 32-key tiles needed (causal)
    for (int kt = 0; kt < nkt; ++kt) {
        const int k0 = kt << 5;

        // ---- scores: preload all 8 K B-fragments, then 8 WMMAs ----
        v16h kf[8];
#pragma unroll
        for (int es = 0; es < 4; ++es) {
            const _Float16* kp = Kb + (size_t)(k0 + lrow) * kDOUT + es * 32 + seg16;
            kf[es] = ldfrag(kp, kp + 8);
            const _Float16* kp1 = kp + 16 * kDOUT;
            kf[es + 4] = ldfrag(kp1, kp1 + 8);
        }
        v8f s0 = {}, s1 = {};
#pragma unroll
        for (int es = 0; es < 4; ++es) {
            s0 = wmma16(qf[es], kf[es], s0);
            s1 = wmma16(qf[es], kf[es + 4], s1);
        }
#pragma unroll
        for (int r = 0; r < 8; ++r) { s0[r] *= SCL; s1[r] *= SCL; }

        // ---- causal mask (only possible on the last tile) ----
        if (kt == nkt - 1) {
#pragma unroll
            for (int r = 0; r < 8; ++r) {
                int q = q0 + rhi + r;
                if (k0 + lrow > q)      s0[r] = -1e30f;
                if (k0 + 16 + lrow > q) s1[r] = -1e30f;
            }
        }

        // ---- online softmax (rows live per 16-lane half, per VGPR) ----
        float al[8];
#pragma unroll
        for (int r = 0; r < 8; ++r) {
            float t = fmaxf(s0[r], s1[r]);
            t = fmaxf(t, __shfl_xor(t, 1));
            t = fmaxf(t, __shfl_xor(t, 2));
            t = fmaxf(t, __shfl_xor(t, 4));
            t = fmaxf(t, __shfl_xor(t, 8));
            float mn = fmaxf(m_run[r], t);
            float a  = exp2f(m_run[r] - mn);
            m_run[r] = mn;
            s0[r] = exp2f(s0[r] - mn);
            s1[r] = exp2f(s1[r] - mn);
            float rs = s0[r] + s1[r];
            rs += __shfl_xor(rs, 1);
            rs += __shfl_xor(rs, 2);
            rs += __shfl_xor(rs, 4);
            rs += __shfl_xor(rs, 8);
            l_run[r] = l_run[r] * a + rs;
            al[r] = a;
        }
#pragma unroll
        for (int nt = 0; nt < 8; ++nt)
#pragma unroll
            for (int r = 0; r < 8; ++r) o[nt][r] *= al[r];

        // ---- transpose P (C layout -> A layout) through per-wave LDS ----
        _Float16 (*pl)[40] = Pl[wave];
#pragma unroll
        for (int r = 0; r < 8; ++r) {
            pl[rhi + r][lrow]      = (_Float16)s0[r];
            pl[rhi + r][lrow + 16] = (_Float16)s1[r];
        }
        asm volatile("s_wait_dscnt 0" ::: "memory");
        const _Float16* pr = &pl[lrow][0] + seg8;
        v16h pf = ldfrag(pr, pr + 8);

        // ---- O += P @ V: preload all 8 V B-fragments, then 8 WMMAs ----
        v16h vf[8];
#pragma unroll
        for (int nt = 0; nt < 8; ++nt) {
            const _Float16* vp = Vb + (size_t)(nt * 16 + lrow) * kS + k0 + seg16;
            vf[nt] = ldfrag(vp, vp + 8);
        }
#pragma unroll
        for (int nt = 0; nt < 8; ++nt)
            o[nt] = wmma16(pf, vf[nt], o[nt]);
    }

    // ---- normalize and store f32 output ----
    float inv[8];
#pragma unroll
    for (int r = 0; r < 8; ++r) inv[r] = 1.0f / l_run[r];
    float* Ob = Out + ((size_t)b * kS + q0) * kDOUT;
#pragma unroll
    for (int nt = 0; nt < 8; ++nt)
#pragma unroll
        for (int r = 0; r < 8; ++r)
            Ob[(size_t)(rhi + r) * kDOUT + nt * 16 + lrow] = o[nt][r] * inv[r];
}

// ===========================================================================
extern "C" void kernel_launch(void* const* d_in, const int* in_sizes, int n_in,
                              void* d_out, int out_size, void* d_ws, size_t ws_size,
                              hipStream_t stream) {
    const float* Xk = (const float*)d_in[0];
    const float* Xv = (const float*)d_in[1];
    const float* Xq = (const float*)d_in[2];
    const float* Wk = (const float*)d_in[3];
    const float* Wv = (const float*)d_in[4];
    const float* Wq = (const float*)d_in[5];

    // ws layout: Kh [B,S,128] f16 | Vt [B,128,S] f16 | Qh [B,S,128] f16  (2MB each)
    char* ws = (char*)d_ws;
    _Float16* Kh = (_Float16*)(ws);
    _Float16* Vt = (_Float16*)(ws + (size_t)2 * 1024 * 1024);
    _Float16* Qh = (_Float16*)(ws + (size_t)4 * 1024 * 1024);

    proj_kernel<<<dim3((kB * kS) / BM, 3), 128, 0, stream>>>(
        Xk, Xv, Xq, Wk, Wv, Wq, Kh, Vt, Qh);
    attn_kernel<<<dim3((kB * kS) / 64), 128, 0, stream>>>(
        Qh, Kh, Vt, (float*)d_out);
}
